// MultiHeadedAttention_86517821214821
// MI455X (gfx1250) — matrix operand order repaired and checked by reading, not compiled, not run on hardware
//
#include <hip/hip_runtime.h>
#include <cstdint>

// ---------------------------------------------------------------------------
// MultiHeadedAttention forward for MI455X (gfx1250), wave32 + WMMA bf16 + TDM.
// B=2, S=2048, HID=1024, H=16, D=64.
// Pass 1: Q/K/V projections  (LDS-tiled bf16 WMMA GEMM, bf16 out in [B,H,S,D])
// Pass 2: fused flash-attention; K/V tiles staged via TENSOR_LOAD_TO_LDS
// Pass 3: output projection  (same GEMM, f32 output)
// ---------------------------------------------------------------------------

typedef __attribute__((ext_vector_type(16))) __bf16 v16bf;
typedef __attribute__((ext_vector_type(8)))  __bf16 v8bf;
typedef __attribute__((ext_vector_type(8)))  float  v8f;
typedef __attribute__((ext_vector_type(4)))  unsigned int v4u;
typedef __attribute__((ext_vector_type(8)))  int    v8i;
typedef __attribute__((ext_vector_type(4)))  int    v4i;

static constexpr int B_   = 2;
static constexpr int S_   = 2048;
static constexpr int HID_ = 1024;
static constexpr int H_   = 16;
static constexpr int D_   = 64;

#if __has_builtin(__builtin_amdgcn_tensor_load_to_lds)
#define HAVE_TDM 1
#else
#define HAVE_TDM 0
#endif

// ---------------------------------------------------------------------------
// Fragment layouts per CDNA5 ISA 7.12.2 (wave32):
//  A (16x32 bf16):  lane L(0-15): row M=L, elems 0-7 -> K+0..7, 8-15 -> K+16..23
//                   lane L+16  : row M=L, elems 0-7 -> K+8..15, 8-15 -> K+24..31
//  B (32x16 bf16):  lane L: col N=L%16, K order as A (elems 0-7 -> K+8h+0..7, 8-15 -> K+16+8h+0..7, h=L/16)
//  C/D (16x16 f32): VGPR r: lanes 0-15 row M=r, lanes 16-31 row M=r+8; N=lane%16
// ---------------------------------------------------------------------------

__device__ inline v16bf load_a_f32(const float* __restrict__ row, int kk, int hi) {
  const int off = kk + (hi ? 8 : 0);
  v16bf f;
#pragma unroll
  for (int i = 0; i < 8; ++i) {
    f[i]     = (__bf16)row[off + i];
    f[i + 8] = (__bf16)row[off + 16 + i];
  }
  return f;
}

// ---------------------------------------------------------------------------
// TDM: load a 32-row x 64-col bf16 tile (contiguous rows, stride 64 elems)
// from global memory into LDS, padding each 32-dword (64 bf16) row with
// 4 dwords (8 bf16) -> matches __shared__ rows of 72 bf16.
// D# layout per CDNA5 ISA ch.8 (groups 0/1; groups 2/3 unused for 2D).
// ---------------------------------------------------------------------------
#if HAVE_TDM
__device__ inline void tdm_load_tile_32x64_bf16(const void* gsrc, unsigned lds_off) {
  const unsigned long long ga = (unsigned long long)(uintptr_t)gsrc;
  v4u g0;
  g0[0] = 1u;                                   // count=1, user descriptor
  g0[1] = lds_off;                              // lds_addr
  g0[2] = (unsigned)ga;                         // global_addr[31:0]
  g0[3] = (unsigned)(ga >> 32) | (2u << 30);    // global_addr[56:32] | type=2
  v8i g1;
  g1[0] = (int)((1u << 16)                      // data_size = 1 -> 2 bytes
              | (1u << 20)                      // pad_enable
              | (4u << 22)                      // pad_interval: 32 dwords
              | (3u << 25));                    // pad_amount:   4 dwords
  g1[1] = (int)(64u << 16);                     // tensor_dim0 = 64
  g1[2] = (int)(32u << 16);                     // tensor_dim1 = 32
  g1[3] = (int)(64u << 16);                     // tile_dim0   = 64
  g1[4] = 32;                                   // tile_dim1   = 32
  g1[5] = 64;                                   // tensor_dim0_stride = 64
  g1[6] = 0;
  g1[7] = 0;
  v4i z4; z4[0] = 0; z4[1] = 0; z4[2] = 0; z4[3] = 0;
#if defined(__clang_major__) && (__clang_major__ >= 23)
  v8i z8; z8[0]=0; z8[1]=0; z8[2]=0; z8[3]=0; z8[4]=0; z8[5]=0; z8[6]=0; z8[7]=0;
  __builtin_amdgcn_tensor_load_to_lds(g0, g1, z4, z4, z8, 0);
#else
  __builtin_amdgcn_tensor_load_to_lds(g0, g1, z4, z4, 0);
#endif
}
#endif

// ---------------------------------------------------------------------------
// Passes 1 & 3: Y = X @ W^T + b.  Block = 8 waves -> 128(M) x 64(N) tile.
// W slab (64 x 32) staged in LDS per k-step, reused by all 8 waves; each
// wave holds 4 accumulators (16 x 64).  TO_BHSD: bf16 out scattered to
// [B,H,S,D]; else f32 row-major [M, HID].
// Grid: (HID/64, (B*S)/128) = (16, 32), block 256.
// ---------------------------------------------------------------------------
template <bool TO_BHSD>
__global__ __launch_bounds__(256) void proj_kernel(
    const float* __restrict__ X, const float* __restrict__ W,
    const float* __restrict__ bias, void* __restrict__ OutP) {
  __shared__ __bf16 Ws[64][40];                 // [n_local][k_local + pad]

  const int lane = threadIdx.x & 31;
  const int w    = threadIdx.x >> 5;
  const int hi   = (lane & 16) ? 1 : 0;
  const int col  = lane & 15;

  const int nbase = blockIdx.x * 64;
  const int mbase = blockIdx.y * 128 + w * 16;
  const float* xrow = X + (size_t)(mbase + col) * HID_;

  v8f acc[4];
#pragma unroll
  for (int n = 0; n < 4; ++n)
#pragma unroll
    for (int r = 0; r < 8; ++r) acc[n][r] = 0.0f;

#pragma unroll 1
  for (int kk = 0; kk < HID_; kk += 32) {
    // cooperative stage of W[nbase..+63][kk..+31] -> bf16 LDS (8 elems/thread)
    {
      const int idx = threadIdx.x * 8;
      const int row = idx >> 5, c = idx & 31;
      const float* src = W + (size_t)(nbase + row) * HID_ + kk + c;
#pragma unroll
      for (int i = 0; i < 8; ++i) Ws[row][c + i] = (__bf16)src[i];
    }
    __syncthreads();

    v16bf a = load_a_f32(xrow, kk, hi);
    const int off = hi ? 8 : 0;
#pragma unroll
    for (int nsub = 0; nsub < 4; ++nsub) {
      const __bf16* wr = &Ws[nsub * 16 + col][0];
      v16bf b;
#pragma unroll
      for (int i = 0; i < 16; ++i) b[i] = wr[off + (i < 8 ? i : i + 8)];
      acc[nsub] = __builtin_amdgcn_wmma_f32_16x16x32_bf16(
          false, a, false, b, (short)0, acc[nsub], false, false);
    }
    __syncthreads();
  }

#pragma unroll
  for (int nsub = 0; nsub < 4; ++nsub) {
    const int n  = nbase + nsub * 16 + col;
    const float bn = bias[n];
#pragma unroll
    for (int r = 0; r < 8; ++r) {
      const int m = mbase + (hi ? r + 8 : r);
      if (TO_BHSD) {
        const int bb = m >> 11, s = m & (S_ - 1);
        const int h = n >> 6, d = n & 63;
        ((__bf16*)OutP)[(((size_t)bb * H_ + h) * S_ + s) * D_ + d] =
            (__bf16)(acc[nsub][r] + bn);
      } else {
        ((float*)OutP)[(size_t)m * HID_ + n] = acc[nsub][r] + bn;
      }
    }
  }
}

// ---------------------------------------------------------------------------
// Pass 2: fused attention. Block = 8 waves = 128 q rows of one (b,h).
// K/V tiles (32 x 64 bf16, rows padded to 72) staged via TDM (wave 0 issues,
// TENSORcnt wait, block barrier). P transposed via per-wave LDS.
// scores = (Q.K^T + bias) * 1/8 + mask ; online softmax ; O += P.V
// Grid: (S/128, B*H), block 256.
// ---------------------------------------------------------------------------
__global__ __launch_bounds__(256) void attn_kernel(
    const __bf16* __restrict__ Qb, const __bf16* __restrict__ Kb,
    const __bf16* __restrict__ Vb, const float* __restrict__ maskp,
    const float* __restrict__ pb, float* __restrict__ Ctx) {
  __shared__ __bf16 Ks[32][72];        // 144B rows (32 data + 4 pad dwords)
  __shared__ __bf16 Vs[32][72];
  __shared__ __bf16 Ps[8][16][40];     // per-wave P scratch (16 x 32 + pad)

  const int lane = threadIdx.x & 31;
  const int w    = threadIdx.x >> 5;
  const int hi   = (lane & 16) ? 1 : 0;
  const int col  = lane & 15;

  const int bh = blockIdx.y;
  const int b  = bh >> 4;
  const int h  = bh & 15;
  const int qbase = blockIdx.x * 128 + w * 16;

  // Q fragments for this wave's 16 rows, K-dim split d=[0,32) and [32,64).
  v16bf aq0, aq1;
  {
    const __bf16* qrow = Qb + ((size_t)bh * S_ + qbase + col) * D_;
    const int off = hi ? 8 : 0;
#pragma unroll
    for (int i = 0; i < 8; ++i) {
      aq0[i]     = qrow[off + i];
      aq0[i + 8] = qrow[off + 16 + i];
      aq1[i]     = qrow[32 + off + i];
      aq1[i + 8] = qrow[32 + off + 16 + i];
    }
  }

  float m_run[8], l_run[8];
  v8f o[4];
#pragma unroll
  for (int r = 0; r < 8; ++r) {
    m_run[r] = -3.0e38f;
    l_run[r] = 0.0f;
#pragma unroll
    for (int dt = 0; dt < 4; ++dt) o[dt][r] = 0.0f;
  }

  const size_t kvbase = (size_t)bh * S_ * D_;

  for (int kk = 0; kk < S_; kk += 32) {
    // -- K/V tile -> LDS --------------------------------------------------
#if HAVE_TDM
    if (w == 0) {
      tdm_load_tile_32x64_bf16(Kb + kvbase + (size_t)kk * D_,
                               (unsigned)(uintptr_t)&Ks[0][0]);
      tdm_load_tile_32x64_bf16(Vb + kvbase + (size_t)kk * D_,
                               (unsigned)(uintptr_t)&Vs[0][0]);
      __builtin_amdgcn_s_wait_tensorcnt(0);
    }
#else
    {
      const int idx = threadIdx.x * 8;
      const int key = idx >> 6, dd = idx & 63;
      const size_t src = kvbase + (size_t)(kk + key) * D_ + dd;
      *reinterpret_cast<v8bf*>(&Ks[key][dd]) =
          *reinterpret_cast<const v8bf*>(Kb + src);
      *reinterpret_cast<v8bf*>(&Vs[key][dd]) =
          *reinterpret_cast<const v8bf*>(Vb + src);
    }
#endif
    __syncthreads();

    // prefetch next bias tile lines (global_prefetch_b8)
    if (kk + 32 < S_)
      __builtin_prefetch(&pb[((size_t)h * S_ + qbase + col) * S_ + kk + 32], 0, 1);

    // -- scores for 2 key sub-tiles of 16 ---------------------------------
    v8f sf[2];
#pragma unroll
    for (int j = 0; j < 2; ++j) {
      const __bf16* krow = &Ks[j * 16 + col][0];
      const int off = hi ? 8 : 0;
      v16bf bk0, bk1;
#pragma unroll
      for (int i = 0; i < 16; ++i) {
        bk0[i] = krow[off + (i < 8 ? i : i + 8)];        // d in [0,32)
        bk1[i] = krow[32 + off + (i < 8 ? i : i + 8)];   // d in [32,64)
      }
      v8f s;
#pragma unroll
      for (int r = 0; r < 8; ++r) s[r] = 0.0f;
      s = __builtin_amdgcn_wmma_f32_16x16x32_bf16(false, aq0, false, bk0,
                                                  (short)0, s, false, false);
      s = __builtin_amdgcn_wmma_f32_16x16x32_bf16(false, aq1, false, bk1,
                                                  (short)0, s, false, false);
#pragma unroll
      for (int r = 0; r < 8; ++r) {
        const int qm = qbase + (hi ? r + 8 : r);
        const int kn = kk + j * 16 + col;
        const float pbv = pb[((size_t)h * S_ + qm) * S_ + kn];
        const float mkv = maskp[((size_t)b * S_ + qm) * S_ + kn];
        s[r] = (s[r] + pbv) * 0.125f + mkv;
      }
      sf[j] = s;
    }

    // -- online softmax over the 32 new columns ---------------------------
    float pexp0[8], pexp1[8];
#pragma unroll
    for (int r = 0; r < 8; ++r) {
      float mx = fmaxf(sf[0][r], sf[1][r]);
#pragma unroll
      for (int sh = 8; sh >= 1; sh >>= 1)
        mx = fmaxf(mx, __shfl_xor(mx, sh, 32));
      const float m_new = fmaxf(m_run[r], mx);
      const float alpha = __expf(m_run[r] - m_new);
      const float p0 = __expf(sf[0][r] - m_new);
      const float p1 = __expf(sf[1][r] - m_new);
      float ssum = p0 + p1;
#pragma unroll
      for (int sh = 8; sh >= 1; sh >>= 1)
        ssum += __shfl_xor(ssum, sh, 32);
      l_run[r] = l_run[r] * alpha + ssum;
      m_run[r] = m_new;
#pragma unroll
      for (int dt = 0; dt < 4; ++dt) o[dt][r] *= alpha;
      pexp0[r] = p0;
      pexp1[r] = p1;
    }

    // -- transpose P (C-layout -> A-layout) through per-wave LDS ----------
#pragma unroll
    for (int r = 0; r < 8; ++r) {
      const int row = hi ? r + 8 : r;
      Ps[w][row][col]      = (__bf16)pexp0[r];
      Ps[w][row][16 + col] = (__bf16)pexp1[r];
    }
    asm volatile("s_wait_dscnt 0" ::: "memory");
    v16bf ap;
    {
      const __bf16* prow = &Ps[w][col][0];
      const int off = hi ? 8 : 0;
#pragma unroll
      for (int i = 0; i < 8; ++i) {
        ap[i]     = prow[off + i];
        ap[i + 8] = prow[off + 16 + i];
      }
    }

    // -- O += P @ V over 4 d-tiles ----------------------------------------
#pragma unroll
    for (int dt = 0; dt < 4; ++dt) {
      const int dcol = dt * 16 + col;
      const int koff = hi ? 8 : 0;
      v16bf bv;
#pragma unroll
      for (int i = 0; i < 16; ++i) bv[i] = Vs[koff + (i < 8 ? i : i + 8)][dcol];
      o[dt] = __builtin_amdgcn_wmma_f32_16x16x32_bf16(false, ap, false, bv,
                                                      (short)0, o[dt], false, false);
    }
    __syncthreads();
  }

  // -- epilogue: O / l -> Ctx in [B, S, H, D] (== [B, S, HID]) ------------
#pragma unroll
  for (int r = 0; r < 8; ++r) {
    const int qm = qbase + (hi ? r + 8 : r);
    const float inv = 1.0f / l_run[r];
#pragma unroll
    for (int dt = 0; dt < 4; ++dt) {
      Ctx[(((size_t)b * S_ + qm) * H_ + h) * D_ + dt * 16 + col] =
          o[dt][r] * inv;
    }
  }
}

// ---------------------------------------------------------------------------
extern "C" void kernel_launch(void* const* d_in, const int* in_sizes, int n_in,
                              void* d_out, int out_size, void* d_ws,
                              size_t ws_size, hipStream_t stream) {
  (void)in_sizes; (void)n_in; (void)out_size; (void)ws_size;

  const float* query = (const float*)d_in[0];
  const float* key   = (const float*)d_in[1];
  const float* value = (const float*)d_in[2];
  const float* maskp = (const float*)d_in[3];
  const float* pb    = (const float*)d_in[4];
  const float* wq    = (const float*)d_in[5];
  const float* bq    = (const float*)d_in[6];
  const float* wk    = (const float*)d_in[7];
  const float* bk    = (const float*)d_in[8];
  const float* wv    = (const float*)d_in[9];
  const float* bv    = (const float*)d_in[10];
  const float* wo    = (const float*)d_in[11];
  const float* bo    = (const float*)d_in[12];

  const size_t QKV_BYTES = (size_t)B_ * H_ * S_ * D_ * sizeof(__bf16); // 8 MB
  char* wsb = (char*)d_ws;
  __bf16* qb  = (__bf16*)(wsb);
  __bf16* kb  = (__bf16*)(wsb + QKV_BYTES);
  __bf16* vb  = (__bf16*)(wsb + 2 * QKV_BYTES);
  float*  ctx = (float*)(wsb + 3 * QKV_BYTES);                          // 16 MB

  const dim3 blk(256);
  const dim3 gproj(HID_ / 64, (B_ * S_) / 128);   // (16, 32)
  const dim3 gattn(S_ / 128, B_ * H_);            // (16, 32)

  proj_kernel<true><<<gproj, blk, 0, stream>>>(query, wq, bq, qb);
  proj_kernel<true><<<gproj, blk, 0, stream>>>(key,   wk, bk, kb);
  proj_kernel<true><<<gproj, blk, 0, stream>>>(value, wv, bv, vb);
  attn_kernel<<<gattn, blk, 0, stream>>>(qb, kb, vb, maskp, pb, ctx);
  proj_kernel<false><<<gproj, blk, 0, stream>>>(ctx, wo, bo, (float*)d_out);
}
